// MemoryModule_26001732010458
// MI455X (gfx1250) — matrix operand order repaired and checked by reading, not compiled, not run on hardware
//
#include <hip/hip_runtime.h>
#include <hip/hip_bf16.h>

typedef __attribute__((ext_vector_type(16))) __bf16 v16bf;
typedef __attribute__((ext_vector_type(8)))  float  v8f;

#define N_TOK 65536   // B*T = 32*2048
#define DIM   512
#define MEMR  64
#define SHRINKC 0.0025f
#define EPSC    1e-12f

union Frag16 { v16bf v; uint32_t u[8]; __bf16 e[16]; };

#define WMMA_BF16(c, a, b) \
  c = __builtin_amdgcn_wmma_f32_16x16x32_bf16(false, (a), false, (b), (short)0, (c), false, false)

// ---- CDNA5 async global->LDS copy (ASYNCcnt), 16 bytes per lane ----
__device__ inline void async_b128(uint32_t lds_byte_off, const void* gaddr) {
  asm volatile("global_load_async_to_lds_b128 %0, %1, off"
               :: "v"(lds_byte_off), "v"((uint64_t)(uintptr_t)gaddr)
               : "memory");
}
__device__ inline void wait_async0() {
  asm volatile("s_wait_asynccnt 0x0" ::: "memory");
}

// A fragment (16x32 bf16, MxK) from row-major bf16, row stride ld.
// Layout: lane<16 holds M=lane, K in {0..15}; lane>=16 holds K in {16..31}.
__device__ inline v16bf ldA_bf16(const __bf16* base, int ld, int m0, int k0, int lane) {
  Frag16 r;
  const __bf16* row = base + (m0 + (lane & 15)) * ld + k0;
  int kh = (lane >> 4) << 4;
#pragma unroll
  for (int j = 0; j < 8; ++j) {
    int kb = kh + (j << 1);
    r.u[j] = *(const uint32_t*)(row + kb);
  }
  return r.v;
}

// B fragment (32x16 bf16, KxN) from a TRANSPOSED [n][k] bf16 array, row stride ld.
__device__ inline v16bf ldB_bf16T(const __bf16* baseT, int ld, int n0, int k0, int lane) {
  Frag16 r;
  const __bf16* row = baseT + (n0 + (lane & 15)) * ld + k0 + ((lane >> 4) << 4);
#pragma unroll
  for (int j = 0; j < 8; ++j) r.u[j] = *(const uint32_t*)(row + (j << 1));
  return r.v;
}

// ---------------------------------------------------------------- prep
__global__ __launch_bounds__(256) void k_prep(const float* __restrict__ memp,
    __bf16* __restrict__ memh, __bf16* __restrict__ meml,
    __bf16* __restrict__ memhT, __bf16* __restrict__ memlT,
    float* __restrict__ addmem) {
  int e = blockIdx.x * 256 + threadIdx.x;          // 0..32767
  int i = e >> 9, j = e & 511;
  float x = memp[e];
  __bf16 h = (__bf16)x;
  __bf16 l = (__bf16)(x - (float)h);
  memh[e] = h; meml[e] = l;
  memhT[j * MEMR + i] = h; memlT[j * MEMR + i] = l;
  addmem[e] = 0.0f;
}

// ---------------------------------------------------------------- S = mem @ q^T  [64, N]
__global__ __launch_bounds__(256) void k_scores(const float* __restrict__ q,
    const __bf16* __restrict__ memh, const __bf16* __restrict__ meml,
    float* __restrict__ S) {
  __shared__ float qlds[32 * 516];                 // 516-f32 rows: conflict-free, 2064B = 129*16
  int tid = threadIdx.x;
  int n0 = blockIdx.x * 32;
  uint32_t ldsbase = (uint32_t)(uintptr_t)qlds;
  // async stage: 32 rows x 512 f32 = 4096 x b128
#pragma unroll 4
  for (int idx4 = tid; idx4 < 4096; idx4 += 256) {
    int r = idx4 >> 7, c4 = idx4 & 127;            // 128 b128-chunks per row
    async_b128(ldsbase + (uint32_t)(r * 516 + c4 * 4) * 4,
               q + (size_t)(n0 + r) * DIM + c4 * 4);
  }
  wait_async0();
  __syncthreads();
  int w = tid >> 5, lane = tid & 31;
  int mt = w & 3, ns = w >> 2;                     // 4 m-tiles x 2 n-subtiles = 8 waves
  v8f acc = {};
  const float* qrow = qlds + (ns * 16 + (lane & 15)) * 516 + ((lane >> 4) << 4);
#pragma unroll 2
  for (int ks = 0; ks < 16; ++ks) {                // K = 512 = 16 * 32
    int k0 = ks * 32;
    v16bf ah = ldA_bf16(memh, DIM, mt * 16, k0, lane);
    v16bf al = ldA_bf16(meml, DIM, mt * 16, k0, lane);
    Frag16 bh, bl;
#pragma unroll
    for (int i = 0; i < 16; ++i) {                 // f32 -> split bf16 on the fly
      float x = qrow[k0 + i];
      __bf16 xh = (__bf16)x;
      bh.e[i] = xh; bl.e[i] = (__bf16)(x - (float)xh);
    }
    WMMA_BF16(acc, ah, bh.v);
    WMMA_BF16(acc, ah, bl.v);
    WMMA_BF16(acc, al, bh.v);
  }
  int ncol = lane & 15, mh = (lane >> 4) << 3;
  size_t ng = (size_t)n0 + ns * 16 + ncol;
#pragma unroll
  for (int j = 0; j < 8; ++j)                      // S stays regular-temporal (L2-resident, re-read twice)
    S[(size_t)(mt * 16 + mh + j) * N_TOK + ng] = acc[j];
}

// ------------------------------------------- per-row softmax stats + attn_u (split bf16)
__global__ __launch_bounds__(256) void k_rowstats(const float* __restrict__ S,
    __bf16* __restrict__ AUh, __bf16* __restrict__ AUl) {
  __shared__ float red[256];
  int tid = threadIdx.x;
  int m = blockIdx.x;
  const float* row = S + (size_t)m * N_TOK;
  float mx = -3.4e38f;
  for (int n = tid; n < N_TOK; n += 256) mx = fmaxf(mx, row[n]);
  red[tid] = mx; __syncthreads();
  for (int s = 128; s > 0; s >>= 1) { if (tid < s) red[tid] = fmaxf(red[tid], red[tid + s]); __syncthreads(); }
  mx = red[0]; __syncthreads();
  float sum = 0.0f;
  for (int n = tid; n < N_TOK; n += 256) sum += __expf(row[n] - mx);
  red[tid] = sum; __syncthreads();
  for (int s = 128; s > 0; s >>= 1) { if (tid < s) red[tid] += red[tid + s]; __syncthreads(); }
  float rinv = 1.0f / red[0]; __syncthreads();
  float l1 = 0.0f;
  for (int n = tid; n < N_TOK; n += 256) {
    float p = __expf(row[n] - mx) * rinv;
    float sv = p - SHRINKC;
    l1 += fmaxf(sv, 0.0f) * p / (fabsf(sv) + EPSC);   // hard_shrink_relu >= 0
  }
  red[tid] = l1; __syncthreads();
  for (int s = 128; s > 0; s >>= 1) { if (tid < s) red[tid] += red[tid + s]; __syncthreads(); }
  float l1inv = 1.0f / fmaxf(red[0], EPSC);
  for (int n = tid; n < N_TOK; n += 256) {
    float p = __expf(row[n] - mx) * rinv;
    float sv = p - SHRINKC;
    float a = fmaxf(sv, 0.0f) * p / (fabsf(sv) + EPSC) * l1inv;
    __bf16 h = (__bf16)a;
    AUh[(size_t)m * N_TOK + n] = h;
    AUl[(size_t)m * N_TOK + n] = (__bf16)(a - (float)h);
  }
}

// ---------------------------------------------------------------- add_mem = attn_u @ q  [64, 512]
__global__ __launch_bounds__(256) void k_addmem(const float* __restrict__ q,
    const __bf16* __restrict__ AUh, const __bf16* __restrict__ AUl,
    float* __restrict__ addmem) {
  __shared__ float qs[128 * 68];                   // 128 k-rows x 64 d (padded, 272B = 17*16)
  int tid = threadIdx.x;
  int kch = blockIdx.x >> 3, db = blockIdx.x & 7;  // 64 K-chunks x 8 d-blocks
  int k0g = kch * 1024, d0 = db * 64;
  int w = tid >> 5, lane = tid & 31;
  int mt = w & 3, dp = w >> 2;                     // wave -> (m-tile, d-tile pair)
  int nl = lane & 15, kh16 = (lane >> 4) << 4;
  uint32_t ldsbase = (uint32_t)(uintptr_t)qs;
  v8f acc0 = {}, acc1 = {};
  for (int sub = 0; sub < 8; ++sub) {              // 8 stages of 128 K
    int kb = k0g + sub * 128;
    __syncthreads();
    // async stage: 128 rows x 64 f32 = 2048 x b128
#pragma unroll
    for (int idx4 = tid; idx4 < 2048; idx4 += 256) {
      int r = idx4 >> 4, c4 = idx4 & 15;           // 16 b128-chunks per row
      async_b128(ldsbase + (uint32_t)(r * 68 + c4 * 4) * 4,
                 q + (size_t)(kb + r) * DIM + d0 + c4 * 4);
    }
    wait_async0();
    __syncthreads();
#pragma unroll
    for (int ks = 0; ks < 4; ++ks) {
      int kloc = ks * 32;
      v16bf ah = ldA_bf16(AUh, N_TOK, mt * 16, kb + kloc, lane);
      v16bf al = ldA_bf16(AUl, N_TOK, mt * 16, kb + kloc, lane);
      const float* src = qs + (kloc + kh16) * 68;
#pragma unroll
      for (int t = 0; t < 2; ++t) {
        int dt = dp * 2 + t;
        Frag16 bh, bl;
#pragma unroll
        for (int i = 0; i < 16; ++i) {
          float x = src[i * 68 + dt * 16 + nl];
          __bf16 xh = (__bf16)x;
          bh.e[i] = xh; bl.e[i] = (__bf16)(x - (float)xh);
        }
        if (t == 0) { WMMA_BF16(acc0, ah, bh.v); WMMA_BF16(acc0, ah, bl.v); WMMA_BF16(acc0, al, bh.v); }
        else        { WMMA_BF16(acc1, ah, bh.v); WMMA_BF16(acc1, ah, bl.v); WMMA_BF16(acc1, al, bh.v); }
      }
    }
  }
  int ncol = lane & 15, mh = (lane >> 4) << 3;
#pragma unroll
  for (int j = 0; j < 8; ++j) {
    unsafeAtomicAdd(&addmem[(mt * 16 + mh + j) * DIM + d0 + (dp * 2) * 16 + ncol],     acc0[j]);
    unsafeAtomicAdd(&addmem[(mt * 16 + mh + j) * DIM + d0 + (dp * 2 + 1) * 16 + ncol], acc1[j]);
  }
}

// ------------------- read path: attn_r (softmax over 64) + out = [q | attn_r @ mem]
__global__ __launch_bounds__(128) void k_read(const float* __restrict__ q,
    const float* __restrict__ S,
    const __bf16* __restrict__ memhT, const __bf16* __restrict__ memlT,
    float* __restrict__ out, float* __restrict__ out_attn) {
  __shared__ float  Sblk[64 * 128];
  __shared__ float  attnf[128 * 65];
  __shared__ __bf16 sAh[128 * 66];
  __shared__ __bf16 sAl[128 * 66];
  int tid = threadIdx.x;
  int n0 = blockIdx.x * 128;
  uint32_t ldsbase = (uint32_t)(uintptr_t)Sblk;
  // async stage: 64 rows x 128 f32 = 2048 x b128
#pragma unroll 4
  for (int idx4 = tid; idx4 < 2048; idx4 += 128) {
    int r = idx4 >> 5, c4 = idx4 & 31;             // 32 b128-chunks per row
    async_b128(ldsbase + (uint32_t)(r * 128 + c4 * 4) * 4,
               S + (size_t)r * N_TOK + n0 + c4 * 4);
  }
  wait_async0();
  __syncthreads();
  {
    int c = tid;                                   // one token per thread
    float mx = -3.4e38f;
    for (int m = 0; m < 64; ++m) mx = fmaxf(mx, Sblk[m * 128 + c]);
    float sum = 0.0f;
    for (int m = 0; m < 64; ++m) { float p = __expf(Sblk[m * 128 + c] - mx); attnf[c * 65 + m] = p; sum += p; }
    float rinv = 1.0f / sum;
    float l1 = 0.0f;
    for (int m = 0; m < 64; ++m) {
      float p = attnf[c * 65 + m] * rinv;
      float sv = p - SHRINKC;
      float h = fmaxf(sv, 0.0f) * p / (fabsf(sv) + EPSC);
      attnf[c * 65 + m] = h; l1 += h;
    }
    float l1inv = 1.0f / fmaxf(l1, EPSC);
    for (int m = 0; m < 64; ++m) {
      float a = attnf[c * 65 + m] * l1inv;
      attnf[c * 65 + m] = a;
      __bf16 h = (__bf16)a;
      sAh[c * 66 + m] = h; sAl[c * 66 + m] = (__bf16)(a - (float)h);
    }
  }
  __syncthreads();
  for (int idx = tid; idx < 128 * 64; idx += 128) {      // attn out (coalesced, streamed -> NT)
    int r = idx >> 6, m = idx & 63;
    __builtin_nontemporal_store(attnf[r * 65 + m], &out_attn[(size_t)(n0 + r) * MEMR + m]);
  }
  for (int idx = tid; idx < 128 * 512; idx += 128) {     // out[:, 0:512] = q (NT stores)
    int r = idx >> 9, c = idx & 511;
    __builtin_nontemporal_store(q[(size_t)(n0 + r) * DIM + c],
                                &out[(size_t)(n0 + r) * (2 * DIM) + c]);
  }
  int w = tid >> 5, lane = tid & 31;
  int ncol = lane & 15, mh = (lane >> 4) << 3;
  for (int dt = 0; dt < 32; ++dt) {                      // 512 d = 32 tiles
    v16bf bh0 = ldB_bf16T(memhT, MEMR, dt * 16, 0,  lane);
    v16bf bl0 = ldB_bf16T(memlT, MEMR, dt * 16, 0,  lane);
    v16bf bh1 = ldB_bf16T(memhT, MEMR, dt * 16, 32, lane);
    v16bf bl1 = ldB_bf16T(memlT, MEMR, dt * 16, 32, lane);
#pragma unroll
    for (int t = 0; t < 2; ++t) {
      int nt = w * 2 + t;                                // 4 waves x 2 = 8 n-tiles
      v16bf a_h0 = ldA_bf16(sAh, 66, nt * 16, 0,  lane);
      v16bf a_l0 = ldA_bf16(sAl, 66, nt * 16, 0,  lane);
      v16bf a_h1 = ldA_bf16(sAh, 66, nt * 16, 32, lane);
      v16bf a_l1 = ldA_bf16(sAl, 66, nt * 16, 32, lane);
      v8f acc = {};
      WMMA_BF16(acc, a_h0, bh0); WMMA_BF16(acc, a_h0, bl0); WMMA_BF16(acc, a_l0, bh0);
      WMMA_BF16(acc, a_h1, bh1); WMMA_BF16(acc, a_h1, bl1); WMMA_BF16(acc, a_l1, bh1);
#pragma unroll
      for (int j = 0; j < 8; ++j)
        __builtin_nontemporal_store(acc[j],
            &out[(size_t)(n0 + nt * 16 + mh + j) * (2 * DIM) + DIM + dt * 16 + ncol]);
    }
  }
}

// ---------------------------------------------------------------- gate + new_mem
__global__ __launch_bounds__(256) void k_gate(const float* __restrict__ memp,
    const float* __restrict__ Uw, const float* __restrict__ Ub,
    const float* __restrict__ Ww, const float* __restrict__ Wb,
    const float* __restrict__ addmem, float* __restrict__ newmem) {
  int e = blockIdx.x * 256 + threadIdx.x;          // 0..32767
  int i = e >> 9, j = e & 511;
  float acc = Ub[j] + Wb[j];
  const float* mr = memp + i * DIM;
  const float* ar = addmem + i * DIM;
  const float* ur = Uw + j * DIM;
  const float* wr = Ww + j * DIM;
  for (int k = 0; k < DIM; ++k) acc += mr[k] * ur[k] + ar[k] * wr[k];
  float g = 1.0f / (1.0f + __expf(-acc));
  newmem[e] = (1.0f - g) * mr[j] + g * ar[j];
}

extern "C" void kernel_launch(void* const* d_in, const int* in_sizes, int n_in,
                              void* d_out, int out_size, void* d_ws, size_t ws_size,
                              hipStream_t stream) {
  (void)in_sizes; (void)n_in; (void)out_size; (void)ws_size;
  const float* q    = (const float*)d_in[0];   // [65536, 512]
  const float* memp = (const float*)d_in[1];   // [64, 512]
  const float* Uw   = (const float*)d_in[2];   // [512, 512]
  const float* Ub   = (const float*)d_in[3];   // [512]
  const float* Ww   = (const float*)d_in[4];   // [512, 512]
  const float* Wb   = (const float*)d_in[5];   // [512]

  char* ws = (char*)d_ws;
  float*  S      = (float*)(ws);                                   // 16 MiB
  __bf16* AUh    = (__bf16*)(ws + (size_t)16777216);               // 8 MiB
  __bf16* AUl    = (__bf16*)(ws + (size_t)25165824);               // 8 MiB
  __bf16* memh   = (__bf16*)(ws + (size_t)33554432);               // 64 KiB each
  __bf16* meml   = memh  + MEMR * DIM;
  __bf16* memhT  = meml  + MEMR * DIM;
  __bf16* memlT  = memhT + MEMR * DIM;
  float*  addmem = (float*)(ws + (size_t)33554432 + 4 * MEMR * DIM * 2);

  float* out      = (float*)d_out;                                 // [N, 1024]
  float* out_attn = out + (size_t)N_TOK * (2 * DIM);               // [N, 64]
  float* newmem   = out_attn + (size_t)N_TOK * MEMR;               // [64, 512]

  k_prep    <<<128,  256, 0, stream>>>(memp, memh, meml, memhT, memlT, addmem);
  k_scores  <<<2048, 256, 0, stream>>>(q, memh, meml, S);
  k_rowstats<<<64,   256, 0, stream>>>(S, AUh, AUl);
  k_addmem  <<<512,  256, 0, stream>>>(q, AUh, AUl, addmem);
  k_read    <<<512,  128, 0, stream>>>(q, S, memhT, memlT, out, out_attn);
  k_gate    <<<128,  256, 0, stream>>>(memp, Uw, Ub, Ww, Wb, addmem, newmem);
}
